// MHA_58093727646235
// MI455X (gfx1250) — matrix operand order repaired and checked by reading, not compiled, not run on hardware
//
#include <hip/hip_runtime.h>
#include <hip/hip_bf16.h>
#include <cstdint>

typedef __attribute__((ext_vector_type(16))) _Float16 v16h;
typedef __attribute__((ext_vector_type(8)))  float    v8f;

#define BB 4
#define TT 2048
#define CC 1024
#define HH 16
#define DD 64
#define MM (BB * TT)   // 8192 tokens

union AReg {
    uint32_t  u[8];
    v16h      v;
    _Float16  h[16];
};

// A-operand (16x32 f16) element K index for VGPR e, half-wave hi.
// VGPR e: K = 2e,2e+1 (+16 if hi half).
__device__ __forceinline__ int a_kidx(int e, int hi) {
    return 2 * e + (hi ? 16 : 0);
}

__device__ __forceinline__ v8f wmma_f16(const AReg& a, const AReg& b, v8f c) {
    return __builtin_amdgcn_wmma_f32_16x16x32_f16(
        /*neg_a=*/false, a.v, /*neg_b=*/false, b.v,
        /*c_mod=*/(short)0, c, /*reuse_a=*/false, /*reuse_b=*/false);
}

__device__ __forceinline__ void load_a16x32(AReg& a, const _Float16* arow, int hi) {
#pragma unroll
    for (int e = 0; e < 8; ++e)
        a.u[e] = *(const uint32_t*)(arow + a_kidx(e, hi));
}

__device__ __forceinline__ void load_b32x16(AReg& b, const _Float16* brow) {
    // brow already includes koff; contiguous 16 halves for this lane.
#pragma unroll
    for (int e = 0; e < 8; ++e)
        b.u[e] = *(const uint32_t*)(brow + 2 * e);
}

// ---------------- conversion kernels ----------------

__global__ void cvt_f32_f16(const float* __restrict__ in,
                            _Float16* __restrict__ out, int n) {
    int i = blockIdx.x * 256 + threadIdx.x;
    if (i < n) out[i] = (_Float16)in[i];
}

// W is [K][N] row-major f32; write WT as [N][K] f16 so WMMA B-operand
// loads (fixed col n, consecutive K) are contiguous dword loads.
__global__ void cvt_w_transpose(const float* __restrict__ in,
                                _Float16* __restrict__ out) {
    int i = blockIdx.x * 256 + threadIdx.x;   // over CC*CC
    int n = i >> 10;
    int k = i & (CC - 1);
    out[(size_t)n * CC + k] = (_Float16)in[(size_t)k * CC + n];
}

// ---------------- QKV projection GEMM ----------------
// Each wave computes a 16(M) x 64(N) block. All 5 operand fetches (A + 4xB)
// are issued before the 4-WMMA burst so loads clause together and overlap
// with the previous iteration's WMMAs.
// vmode 0: store as [b,h,t,d]; vmode 1: store V transposed [b,h,d,t].
__global__ __launch_bounds__(256) void proj_gemm(
        const _Float16* __restrict__ A, const _Float16* __restrict__ Bt,
        const float* __restrict__ bias, _Float16* __restrict__ out, int vmode) {
    int lane = threadIdx.x & 31, w = threadIdx.x >> 5;
    int hi = (lane >> 4) & 1, ln = lane & 15;
    int tid = blockIdx.x * 8 + w;          // 8192 wave-tiles
    int ng    = tid & (CC / 64 - 1);       // 16 col groups of 64
    int tileM = tid >> 4;                  // 512 row tiles
    int row = tileM * 16 + ln;
    int nb  = ng * 64;
    int koff = hi ? 16 : 0;

    const _Float16* arow0 = A + (size_t)row * CC;
    const _Float16* bcol0 = Bt + (size_t)(nb + ln) * CC + koff;

    v8f c0 = {}, c1 = {}, c2 = {}, c3 = {};
    for (int k0 = 0; k0 < CC; k0 += 32) {
        AReg a, b0, b1, b2, b3;
        load_a16x32(a, arow0 + k0, hi);
        load_b32x16(b0, bcol0 + k0 + (size_t)(0 * 16) * CC);
        load_b32x16(b1, bcol0 + k0 + (size_t)(1 * 16) * CC);
        load_b32x16(b2, bcol0 + k0 + (size_t)(2 * 16) * CC);
        load_b32x16(b3, bcol0 + k0 + (size_t)(3 * 16) * CC);
        if (k0 + 32 < CC) __builtin_prefetch(arow0 + k0 + 32, 0, 3);
        c0 = wmma_f16(a, b0, c0);
        c1 = wmma_f16(a, b1, c1);
        c2 = wmma_f16(a, b2, c2);
        c3 = wmma_f16(a, b3, c3);
    }

    v8f cs[4] = {c0, c1, c2, c3};
#pragma unroll
    for (int j = 0; j < 4; ++j) {
        int col = nb + j * 16 + ln;
        float bval = bias[col];
        int h = col / DD, dd = col % DD;
#pragma unroll
        for (int v = 0; v < 8; ++v) {
            float val = cs[j][v] + bval;
            int gm = tileM * 16 + v + 8 * hi;       // global token row
            int bidx = gm / TT, t = gm % TT;
            size_t idx;
            if (vmode == 0)
                idx = (((size_t)(bidx * HH + h)) * TT + t) * DD + dd;
            else
                idx = (((size_t)(bidx * HH + h)) * DD + dd) * TT + t;
            out[idx] = (_Float16)val;
        }
    }
}

// ---------------- flash attention ----------------
// Qh,Kh: [bh][t][d] f16.  Vt: [bh][d][t] f16.  Ah out: [b*t][C] f16.
__global__ __launch_bounds__(256) void flash_attn(
        const _Float16* __restrict__ Qh, const _Float16* __restrict__ Kh,
        const _Float16* __restrict__ Vt, _Float16* __restrict__ Ah) {
    __shared__ _Float16 plds[8][16][32];   // per-wave P staging (C->A transpose)

    int lane = threadIdx.x & 31, w = threadIdx.x >> 5;
    int hi = (lane >> 4) & 1, ln = lane & 15;
    int bh = blockIdx.x >> 4;              // 0..63
    int qblk = blockIdx.x & 15;            // 16 chunks of 128 queries
    int b = bh >> 4, h = bh & 15;
    int q0 = qblk * 128 + w * 16;
    int koff = hi ? 16 : 0;

    const _Float16* qbase = Qh + (size_t)bh * TT * DD;
    const _Float16* kbase = Kh + (size_t)bh * TT * DD;
    const _Float16* vbase = Vt + (size_t)bh * DD * TT;

    // Q tile as two 16x32 A operands (d = 64)
    AReg qa0, qa1;
    load_a16x32(qa0, qbase + (size_t)(q0 + ln) * DD, hi);
    load_a16x32(qa1, qbase + (size_t)(q0 + ln) * DD + 32, hi);

    float mrow[8], lrow[8];
#pragma unroll
    for (int v = 0; v < 8; ++v) { mrow[v] = -1e30f; lrow[v] = 0.f; }
    v8f o0 = {}, o1 = {}, o2 = {}, o3 = {};

    for (int s0 = 0; s0 < TT; s0 += 32) {
        // scores: keys [s0,s0+16) -> sa, [s0+16,s0+32) -> sb
        AReg ka0, ka1, kb0, kb1;
        load_b32x16(ka0, kbase + (size_t)(s0 + ln) * DD + koff);
        load_b32x16(ka1, kbase + (size_t)(s0 + ln) * DD + 32 + koff);
        load_b32x16(kb0, kbase + (size_t)(s0 + 16 + ln) * DD + koff);
        load_b32x16(kb1, kbase + (size_t)(s0 + 16 + ln) * DD + 32 + koff);
        v8f sa = {}, sb = {};
        sa = wmma_f16(qa0, ka0, sa);
        sa = wmma_f16(qa1, ka1, sa);
        sb = wmma_f16(qa0, kb0, sb);
        sb = wmma_f16(qa1, kb1, sb);

        // online softmax; row r = v + 8*hi lives in one 16-lane half
#pragma unroll
        for (int v = 0; v < 8; ++v) {
            float x0 = sa[v] * 0.125f, x1 = sb[v] * 0.125f;  // 1/sqrt(64)
            float tmx = fmaxf(x0, x1);
#pragma unroll
            for (int msk = 1; msk < 16; msk <<= 1)
                tmx = fmaxf(tmx, __shfl_xor(tmx, msk, 32));
            float nm = fmaxf(mrow[v], tmx);
            float alpha = __expf(mrow[v] - nm);
            float e0 = __expf(x0 - nm), e1 = __expf(x1 - nm);
            float rs = e0 + e1;
#pragma unroll
            for (int msk = 1; msk < 16; msk <<= 1)
                rs += __shfl_xor(rs, msk, 32);
            lrow[v] = lrow[v] * alpha + rs;
            mrow[v] = nm;
            o0[v] *= alpha; o1[v] *= alpha; o2[v] *= alpha; o3[v] *= alpha;
            int r = v + 8 * hi;
            plds[w][r][ln]      = (_Float16)e0;
            plds[w][r][ln + 16] = (_Float16)e1;
        }

        // reload P in A-operand layout (16x32)
        AReg pa;
#pragma unroll
        for (int e = 0; e < 8; ++e) {
            int kk = a_kidx(e, hi);
            pa.h[2 * e]     = plds[w][ln][kk];
            pa.h[2 * e + 1] = plds[w][ln][kk + 1];
        }

        // O += P @ V over 4 column groups of d (batch loads, then WMMA burst)
        AReg vb0, vb1, vb2, vb3;
        load_b32x16(vb0, vbase + (size_t)(0 * 16 + ln) * TT + s0 + koff);
        load_b32x16(vb1, vbase + (size_t)(1 * 16 + ln) * TT + s0 + koff);
        load_b32x16(vb2, vbase + (size_t)(2 * 16 + ln) * TT + s0 + koff);
        load_b32x16(vb3, vbase + (size_t)(3 * 16 + ln) * TT + s0 + koff);
        o0 = wmma_f16(pa, vb0, o0);
        o1 = wmma_f16(pa, vb1, o1);
        o2 = wmma_f16(pa, vb2, o2);
        o3 = wmma_f16(pa, vb3, o3);
    }

    // normalize + store to [b*t][C] f16 for the output projection
#pragma unroll
    for (int v = 0; v < 8; ++v) {
        float inv = 1.0f / lrow[v];
        int t = q0 + v + 8 * hi;
        size_t base = ((size_t)(b * TT + t)) * CC + h * DD;
        Ah[base + 0 * 16 + ln] = (_Float16)(o0[v] * inv);
        Ah[base + 1 * 16 + ln] = (_Float16)(o1[v] * inv);
        Ah[base + 2 * 16 + ln] = (_Float16)(o2[v] * inv);
        Ah[base + 3 * 16 + ln] = (_Float16)(o3[v] * inv);
    }
}

// ---------------- output projection GEMM (f32 out) ----------------
// Same 16x64 wave tile / batched-load structure as proj_gemm.
__global__ __launch_bounds__(256) void out_gemm(
        const _Float16* __restrict__ A, const _Float16* __restrict__ Bt,
        const float* __restrict__ bias, float* __restrict__ out) {
    int lane = threadIdx.x & 31, w = threadIdx.x >> 5;
    int hi = (lane >> 4) & 1, ln = lane & 15;
    int tid = blockIdx.x * 8 + w;
    int ng    = tid & (CC / 64 - 1);
    int tileM = tid >> 4;
    int row = tileM * 16 + ln;
    int nb  = ng * 64;
    int koff = hi ? 16 : 0;

    const _Float16* arow0 = A + (size_t)row * CC;
    const _Float16* bcol0 = Bt + (size_t)(nb + ln) * CC + koff;

    v8f c0 = {}, c1 = {}, c2 = {}, c3 = {};
    for (int k0 = 0; k0 < CC; k0 += 32) {
        AReg a, b0, b1, b2, b3;
        load_a16x32(a, arow0 + k0, hi);
        load_b32x16(b0, bcol0 + k0 + (size_t)(0 * 16) * CC);
        load_b32x16(b1, bcol0 + k0 + (size_t)(1 * 16) * CC);
        load_b32x16(b2, bcol0 + k0 + (size_t)(2 * 16) * CC);
        load_b32x16(b3, bcol0 + k0 + (size_t)(3 * 16) * CC);
        if (k0 + 32 < CC) __builtin_prefetch(arow0 + k0 + 32, 0, 3);
        c0 = wmma_f16(a, b0, c0);
        c1 = wmma_f16(a, b1, c1);
        c2 = wmma_f16(a, b2, c2);
        c3 = wmma_f16(a, b3, c3);
    }

    v8f cs[4] = {c0, c1, c2, c3};
#pragma unroll
    for (int j = 0; j < 4; ++j) {
        int col = nb + j * 16 + ln;
        float bval = bias[col];
#pragma unroll
        for (int v = 0; v < 8; ++v) {
            int gm = tileM * 16 + v + 8 * hi;
            out[(size_t)gm * CC + col] = cs[j][v] + bval;
        }
    }
}

// ---------------- host launch ----------------
extern "C" void kernel_launch(void* const* d_in, const int* in_sizes, int n_in,
                              void* d_out, int out_size, void* d_ws, size_t ws_size,
                              hipStream_t stream) {
    (void)in_sizes; (void)n_in; (void)out_size; (void)ws_size;
    const float* x  = (const float*)d_in[0];
    const float* Wq = (const float*)d_in[1];
    const float* bq = (const float*)d_in[2];
    const float* Wk = (const float*)d_in[3];
    const float* bk = (const float*)d_in[4];
    const float* Wv = (const float*)d_in[5];
    const float* bv = (const float*)d_in[6];
    const float* Wo = (const float*)d_in[7];
    const float* bo = (const float*)d_in[8];
    float* out = (float*)d_out;

    char* ws = (char*)d_ws;
    const size_t MB = 1024ull * 1024ull;
    _Float16* Xh  = (_Float16*)(ws + 0);        // 16 MB  [M][C]
    _Float16* WqT = (_Float16*)(ws + 16 * MB);  // 2 MB   [N][K]
    _Float16* WkT = (_Float16*)(ws + 18 * MB);
    _Float16* WvT = (_Float16*)(ws + 20 * MB);
    _Float16* WoT = (_Float16*)(ws + 22 * MB);
    _Float16* Qh  = (_Float16*)(ws + 24 * MB);  // 16 MB  [bh][t][d]
    _Float16* Kh  = (_Float16*)(ws + 40 * MB);  // 16 MB  [bh][t][d]
    _Float16* Vt  = (_Float16*)(ws + 56 * MB);  // 16 MB  [bh][d][t]
    _Float16* Ah  = (_Float16*)(ws + 72 * MB);  // 16 MB  [m][C]

    int nx = MM * CC;
    cvt_f32_f16<<<(nx + 255) / 256, 256, 0, stream>>>(x, Xh, nx);
    int nw = CC * CC;
    int gw = (nw + 255) / 256;
    cvt_w_transpose<<<gw, 256, 0, stream>>>(Wq, WqT);
    cvt_w_transpose<<<gw, 256, 0, stream>>>(Wk, WkT);
    cvt_w_transpose<<<gw, 256, 0, stream>>>(Wv, WvT);
    cvt_w_transpose<<<gw, 256, 0, stream>>>(Wo, WoT);

    dim3 gg((MM / 16) * (CC / 64) / 8);   // 1024 blocks, 8 wave-tiles each
    proj_gemm<<<gg, 256, 0, stream>>>(Xh, WqT, bq, Qh, 0);
    proj_gemm<<<gg, 256, 0, stream>>>(Xh, WkT, bk, Kh, 0);
    proj_gemm<<<gg, 256, 0, stream>>>(Xh, WvT, bv, Vt, 1);

    flash_attn<<<BB * HH * (TT / 128), 256, 0, stream>>>(Qh, Kh, Vt, Ah);

    out_gemm<<<gg, 256, 0, stream>>>(Ah, WoT, bo, out);
}
